// DynamicLinear_71932112274008
// MI455X (gfx1250) — compile-verified, matrix operand order checked by reading, not run on hardware
//
#include <hip/hip_runtime.h>

// ---------------- problem constants (from reference) ----------------
#define B_ROWS  4096
#define IN_DIM  2048
#define OUT_DIM 2048
#define NE      4          // experts

// ---------------- tiling ----------------
#define BM      128        // block rows (batch)
#define BN      64         // block cols (out features)
#define KT      32         // K-chunk per stage (matches 16x16x32 bf16 WMMA)
#define XSTR    40         // LDS row stride in bf16 (32 + 8 pad -> conflict-free b128)
#define WSTR    40
#define NSTEP   (IN_DIM / KT)   // 64

#if defined(__gfx1250__) && __has_builtin(__builtin_amdgcn_global_load_async_to_lds_b128)
#define ASYNC_OK 1
#else
#define ASYNC_OK 0
#endif

typedef __attribute__((ext_vector_type(16))) __bf16 bf16x16;
typedef __attribute__((ext_vector_type(8)))  __bf16 bf16x8;
typedef __attribute__((ext_vector_type(4)))  __bf16 bf16x4;
typedef __attribute__((ext_vector_type(8)))  float  f32x8;
typedef __attribute__((ext_vector_type(4)))  int    i32x4;

// address-space-qualified b128 payload pointers for the async-copy builtin
typedef __attribute__((address_space(1))) i32x4 gas_i32x4;   // global
typedef __attribute__((address_space(3))) i32x4 las_i32x4;   // LDS

union Frag16 { bf16x16 v; bf16x8 h[2]; };

__device__ __forceinline__ void wait_async0() {
#if __has_builtin(__builtin_amdgcn_s_wait_asynccnt)
  __builtin_amdgcn_s_wait_asynccnt(0);
#else
  asm volatile("s_wait_asynccnt 0x0" ::: "memory");
#endif
}

// ---------------- f32 -> bf16 pre-convert ----------------
__global__ void cvt_f32_to_bf16(const float* __restrict__ in,
                                __bf16* __restrict__ out, int n4) {
  int i = blockIdx.x * blockDim.x + threadIdx.x;
  if (i >= n4) return;
  float4 v = ((const float4*)in)[i];
  bf16x4 t;
  t[0] = (__bf16)v.x; t[1] = (__bf16)v.y; t[2] = (__bf16)v.z; t[3] = (__bf16)v.w;
  ((bf16x4*)out)[i] = t;
}

// ---------------- fused 4-expert GEMM + per-row mix + bias ----------------
// out[b,o] = sum_k attn[b,k] * ( (x @ W_k^T)[b,o] + bias[k,o] )
template <bool BF16SRC>
__global__ __launch_bounds__(256)
void moe_wmma_gemm(const void* __restrict__ xsrc_v,
                   const float* __restrict__ attn,
                   const void* __restrict__ wsrc_v,
                   const float* __restrict__ bias,
                   float* __restrict__ out) {
  __shared__ __attribute__((aligned(16))) __bf16 xs[2][BM * XSTR];        // 20 KB
  __shared__ __attribute__((aligned(16))) __bf16 wl[2][NE * BN * WSTR];   // 40 KB

  const int tid  = threadIdx.x;
  const int lane = tid & 31;
  const int wave = tid >> 5;            // 0..7
  const int wm0  = (wave >> 1) * 32;    // 4 waves along M
  const int wn0  = (wave & 1) * 32;     // 2 waves along N
  const int lr   = lane & 15;
  const int half = lane >> 4;
  const int kA   = half * 8;            // lane's first K-chunk inside a 32-deep tile

  const int bn0 = blockIdx.x * BN;
  const int bm0 = blockIdx.y * BM;

  // accumulators: [expert][m-subtile][n-subtile], f32 16x16 tile = 8 VGPRs
  const f32x8 fzero = {0.f,0.f,0.f,0.f,0.f,0.f,0.f,0.f};
  f32x8 acc[NE][2][2];
#pragma unroll
  for (int k = 0; k < NE; ++k)
#pragma unroll
    for (int mt = 0; mt < 2; ++mt)
#pragma unroll
      for (int nt = 0; nt < 2; ++nt) acc[k][mt][nt] = fzero;

  // ------- compute one K-stage from LDS buffer `cur` -------
  auto compute = [&](int cur) {
    Frag16 a[2];
#pragma unroll
    for (int mt = 0; mt < 2; ++mt) {
      const __bf16* ab = &xs[cur][(wm0 + mt * 16 + lr) * XSTR + kA];
      a[mt].h[0] = *(const bf16x8*)(ab);
      a[mt].h[1] = *(const bf16x8*)(ab + 16);
    }
    // rotate two B fragments so each ds_load pair hides under a WMMA pair
    Frag16 bfr[2];
    auto bload = [&](Frag16& f, int j) {
      const int k = j >> 1, nt = j & 1;
      const __bf16* bb = &wl[cur][(k * BN + wn0 + nt * 16 + lr) * WSTR + kA];
      f.h[0] = *(const bf16x8*)(bb);
      f.h[1] = *(const bf16x8*)(bb + 16);
    };
    bload(bfr[0], 0);
#pragma unroll
    for (int j = 0; j < 8; ++j) {
      if (j + 1 < 8) bload(bfr[(j + 1) & 1], j + 1);
      const int k = j >> 1, nt = j & 1;
      acc[k][0][nt] = __builtin_amdgcn_wmma_f32_16x16x32_bf16(
          false, a[0].v, false, bfr[j & 1].v, (short)0, acc[k][0][nt], false, false);
      acc[k][1][nt] = __builtin_amdgcn_wmma_f32_16x16x32_bf16(
          false, a[1].v, false, bfr[j & 1].v, (short)0, acc[k][1][nt], false, false);
    }
  };

#if ASYNC_OK
  if (BF16SRC) {
    // ======== async global->LDS DMA pipeline (no staging VGPRs) ========
    const __bf16* xbf = (const __bf16*)xsrc_v;
    const __bf16* wbf = (const __bf16*)wsrc_v;

    auto issue_async = [&](int buf, int s) {
      const int kk = s * KT;
#pragma unroll
      for (int i = 0; i < 2; ++i) {       // x tile: 128 rows x 4 chunks of 8 bf16
        const int idx = tid + i * 256, row = idx >> 2, c8 = idx & 3;
        const __bf16* g = xbf + (size_t)(bm0 + row) * IN_DIM + kk + c8 * 8;
        __bf16* l = &xs[buf][row * XSTR + c8 * 8];
        __builtin_amdgcn_global_load_async_to_lds_b128(
            (gas_i32x4*)g, (las_i32x4*)l, 0, 0);
      }
#pragma unroll
      for (int i = 0; i < 4; ++i) {       // w tiles: 4 experts x 64 rows x 4 chunks
        const int idx = tid + i * 256, row = idx >> 2, c8 = idx & 3;
        const int k = row >> 6, ol = row & 63;
        const __bf16* g = wbf + ((size_t)k * OUT_DIM + bn0 + ol) * IN_DIM + kk + c8 * 8;
        __bf16* l = &wl[buf][(k * BN + ol) * WSTR + c8 * 8];
        __builtin_amdgcn_global_load_async_to_lds_b128(
            (gas_i32x4*)g, (las_i32x4*)l, 0, 0);
      }
    };

    issue_async(0, 0);
    wait_async0();
    __syncthreads();
    for (int s = 0; s < NSTEP; ++s) {
      if (s + 1 < NSTEP) issue_async((s + 1) & 1, s + 1);
      compute(s & 1);
      if (s + 1 < NSTEP) {
        wait_async0();
        __syncthreads();
      }
    }
  } else
#endif
  {
    // ======== fallback: register-staged, short lifetimes, single buffer ========
    for (int s = 0; s < NSTEP; ++s) {
      const int kk = s * KT;
      if (s) __syncthreads();             // protect buffer 0 from previous compute
      if (BF16SRC) {
        const __bf16* xbf = (const __bf16*)xsrc_v;
        const __bf16* wbf = (const __bf16*)wsrc_v;
#pragma unroll
        for (int i = 0; i < 2; ++i) {
          const int idx = tid + i * 256, row = idx >> 2, c8 = idx & 3;
          uint4 v = *(const uint4*)(xbf + (size_t)(bm0 + row) * IN_DIM + kk + c8 * 8);
          *(uint4*)&xs[0][row * XSTR + c8 * 8] = v;
        }
#pragma unroll
        for (int i = 0; i < 4; ++i) {
          const int idx = tid + i * 256, row = idx >> 2, c8 = idx & 3;
          const int k = row >> 6, ol = row & 63;
          uint4 v = *(const uint4*)(wbf + ((size_t)k * OUT_DIM + bn0 + ol) * IN_DIM + kk + c8 * 8);
          *(uint4*)&wl[0][(k * BN + ol) * WSTR + c8 * 8] = v;
        }
      } else {
        const float* xf = (const float*)xsrc_v;
        const float* wf = (const float*)wsrc_v;
#pragma unroll
        for (int i = 0; i < 4; ++i) {
          const int idx = tid + i * 256, row = idx >> 3, c4 = idx & 7;
          float4 v = *(const float4*)(xf + (size_t)(bm0 + row) * IN_DIM + kk + c4 * 4);
          bf16x4 t;
          t[0] = (__bf16)v.x; t[1] = (__bf16)v.y; t[2] = (__bf16)v.z; t[3] = (__bf16)v.w;
          *(bf16x4*)&xs[0][row * XSTR + c4 * 4] = t;
        }
#pragma unroll
        for (int i = 0; i < 8; ++i) {
          const int idx = tid + i * 256, row = idx >> 3, c4 = idx & 7;
          const int k = row >> 6, ol = row & 63;
          float4 v = *(const float4*)(wf + ((size_t)k * OUT_DIM + bn0 + ol) * IN_DIM + kk + c4 * 4);
          bf16x4 t;
          t[0] = (__bf16)v.x; t[1] = (__bf16)v.y; t[2] = (__bf16)v.z; t[3] = (__bf16)v.w;
          *(bf16x4*)&wl[0][(k * BN + ol) * WSTR + c4 * 4] = t;
        }
      }
      __syncthreads();
      compute(0);
    }
  }

  // ---- epilogue: out[b,o] = sum_k attn[b,k] * (acc_k[b,o] + bias[k,o]) ----
  int gcol[2];
  float bv[NE][2];
#pragma unroll
  for (int nt = 0; nt < 2; ++nt) {
    gcol[nt] = bn0 + wn0 + nt * 16 + lr;          // C/D layout: N = lane%16
#pragma unroll
    for (int k = 0; k < NE; ++k) bv[k][nt] = bias[k * OUT_DIM + gcol[nt]];
  }
#pragma unroll
  for (int mt = 0; mt < 2; ++mt) {
#pragma unroll
    for (int d = 0; d < 8; ++d) {                 // C/D layout: M = d + 8*half
      const int grow = bm0 + wm0 + mt * 16 + half * 8 + d;
      const float4 a4 = *(const float4*)(attn + (size_t)grow * NE);
#pragma unroll
      for (int nt = 0; nt < 2; ++nt) {
        float r = a4.x * (acc[0][mt][nt][d] + bv[0][nt])
                + a4.y * (acc[1][mt][nt][d] + bv[1][nt])
                + a4.z * (acc[2][mt][nt][d] + bv[2][nt])
                + a4.w * (acc[3][mt][nt][d] + bv[3][nt]);
        out[(size_t)grow * OUT_DIM + gcol[nt]] = r;
      }
    }
  }
}

// ---------------- host launch ----------------
extern "C" void kernel_launch(void* const* d_in, const int* in_sizes, int n_in,
                              void* d_out, int out_size, void* d_ws, size_t ws_size,
                              hipStream_t stream) {
  const float* x    = (const float*)d_in[0];   // [4096, 2048]
  const float* attn = (const float*)d_in[1];   // [4096, 4]
  const float* w    = (const float*)d_in[2];   // [4, 2048, 2048]
  const float* bias = (const float*)d_in[3];   // [4, 2048]
  float* out = (float*)d_out;                  // [4096, 2048]

  const size_t xe = (size_t)B_ROWS * IN_DIM;
  const size_t we = (size_t)NE * OUT_DIM * IN_DIM;

  dim3 grid(OUT_DIM / BN, B_ROWS / BM);        // 32 x 32 blocks
  dim3 blk(256);

  if (ws_size >= (xe + we) * sizeof(__bf16)) {
    // pre-convert x and W to bf16 once (L2-resident afterwards)
    __bf16* xbf = (__bf16*)d_ws;
    __bf16* wbf = xbf + xe;
    cvt_f32_to_bf16<<<(unsigned)((xe / 4 + 255) / 256), 256, 0, stream>>>(x, xbf, (int)(xe / 4));
    cvt_f32_to_bf16<<<(unsigned)((we / 4 + 255) / 256), 256, 0, stream>>>(w, wbf, (int)(we / 4));
    moe_wmma_gemm<true><<<grid, blk, 0, stream>>>(xbf, attn, wbf, bias, out);
  } else {
    // fallback: convert f32->bf16 during the global->LDS stage
    moe_wmma_gemm<false><<<grid, blk, 0, stream>>>(x, attn, w, bias, out);
  }
}
